// Discriminator_21655225107210
// MI455X (gfx1250) — matrix operand order repaired and checked by reading, not compiled, not run on hardware
//
#include <hip/hip_runtime.h>
#include <hip/hip_bf16.h>

typedef __bf16 bf16_t;
typedef __attribute__((ext_vector_type(16))) __bf16 v16bf;
typedef __attribute__((ext_vector_type(8)))  __bf16 v8bf;
typedef __attribute__((ext_vector_type(8)))  float  v8f;

// Build a 16-elem bf16 fragment from two contiguous 8-elem (16-byte) runs.
static __device__ __forceinline__ v16bf make_frag(const bf16_t* p0, const bf16_t* p1) {
    v8bf lo = *(const v8bf*)p0;
    v8bf hi = *(const v8bf*)p1;
    return __builtin_shufflevector(lo, hi, 0,1,2,3,4,5,6,7,8,9,10,11,12,13,14,15);
}

// ---------------- conversion kernels ----------------

__global__ void k_cvt_bf16(const float* __restrict__ in, bf16_t* __restrict__ out, int n) {
    int i = blockIdx.x * blockDim.x + threadIdx.x;
    if (i < n) out[i] = (bf16_t)in[i];
}

// W [rows, cols] fp32 -> WT [cols, rows] bf16 (K-contiguous per output column)
__global__ void k_cvt_transpose(const float* __restrict__ in, bf16_t* __restrict__ out,
                                int rows, int cols) {
    int i = blockIdx.x * blockDim.x + threadIdx.x;
    if (i >= rows * cols) return;
    int r = i / cols, c = i % cols;
    out[c * rows + r] = (bf16_t)in[i];
}

// Wm [krows, dout] fp32 -> WmT [dout, kpad] bf16, zero-padded K
__global__ void k_cvt_transpose_pad(const float* __restrict__ in, bf16_t* __restrict__ out,
                                    int krows, int kpad, int dout) {
    int i = blockIdx.x * blockDim.x + threadIdx.x;
    if (i >= dout * kpad) return;
    int n = i / kpad, k = i % kpad;
    out[i] = (k < krows) ? (bf16_t)in[k * dout + n] : (bf16_t)0.0f;
}

__global__ void k_zero_f32(float* __restrict__ p, int n) {
    int i = blockIdx.x * blockDim.x + threadIdx.x;
    if (i < n) p[i] = 0.0f;
}

// ---------------- edge message GEMM + scatter ----------------
// msg = concat(h[src], edge_attr, 0pad) @ Wm ; agg[dst] += msg
// 64 threads = 2 waves; each wave owns a 32-edge tile (2 x 16-row A tiles
// sharing each B fragment -> 2 independent WMMAs per B load pair).
// All global indices are 32-bit so the backend can use SGPR-base + v-offset
// addressing (no 64-bit address math in the scatter).
template<int DIN, int DOUT, int KPAD>
__global__ __launch_bounds__(64)
void k_edge(const bf16_t* __restrict__ h,    // [N, DIN]
            const bf16_t* __restrict__ ea,   // [E, 8]
            const int*    __restrict__ src,
            const int*    __restrict__ dst,
            const bf16_t* __restrict__ wmT,  // [DOUT, KPAD] (transposed, padded)
            float*        __restrict__ agg)  // [N, DOUT]
{
    __shared__ bf16_t sA[2][32][KPAD];
    __shared__ bf16_t sW[DOUT * KPAD];

    const int tid  = threadIdx.x;
    const int wave = tid >> 5;
    const int lane = tid & 31;

    // stage transposed weights once per block (16B vector copies)
    {
        const uint4* wsrcv = (const uint4*)wmT;
        uint4* wdstv = (uint4*)sW;
#pragma unroll 4
        for (int i = tid; i < DOUT * KPAD / 8; i += 64) wdstv[i] = wsrcv[i];
    }

    const int e0 = blockIdx.x * 64 + wave * 32;

    // stage 32 gathered rows: concat(h[src], edge_attr, 0pad), 8B vector copies
    for (int r = 0; r < 32; ++r) {
        const int e = e0 + r;
        const int s = src[e];
        const uint2* hrow = (const uint2*)(h + s * DIN);
        const uint2* erow = (const uint2*)(ea + e * 8);
        uint2* arow = (uint2*)(&sA[wave][r][0]);
#pragma unroll
        for (int c = lane; c < KPAD / 4; c += 32) {
            const int k = c * 4;
            uint2 v;
            if (k < DIN)          v = hrow[c];
            else if (k < DIN + 8) v = erow[(k - DIN) / 4];
            else                  { v.x = 0u; v.y = 0u; }
            arow[c] = v;
        }
    }
    __syncthreads();

    constexpr int KT = KPAD / 32;
    const int mrow = lane & 15;
    const int hi   = lane >> 4;

    // hoisted A fragments for both 16-row subtiles
    v16bf afrag[2][KT];
    const bf16_t* pa0 = &sA[wave][mrow][hi * 8];
    const bf16_t* pa1 = &sA[wave][16 + mrow][hi * 8];
#pragma unroll
    for (int kt = 0; kt < KT; ++kt) {
        afrag[0][kt] = make_frag(pa0 + kt * 32, pa0 + kt * 32 + 16);
        afrag[1][kt] = make_frag(pa1 + kt * 32, pa1 + kt * 32 + 16);
    }

    // hoisted 32-bit scatter offsets (element index into agg)
    int doff0[8], doff1[8];
#pragma unroll
    for (int r = 0; r < 8; ++r) {
        doff0[r] = dst[e0 + r + hi * 8] * DOUT + mrow;
        doff1[r] = dst[e0 + 16 + r + hi * 8] * DOUT + mrow;
    }

#pragma unroll
    for (int nt = 0; nt < DOUT / 16; ++nt) {
        const int ncol = nt * 16 + mrow;
        const bf16_t* pb = &sW[ncol * KPAD + hi * 8];
        v8f acc0 = {}, acc1 = {};
#pragma unroll
        for (int kt = 0; kt < KT; ++kt) {
            v16bf b = make_frag(pb + kt * 32, pb + kt * 32 + 16);
            acc0 = __builtin_amdgcn_wmma_f32_16x16x32_bf16(
                false, afrag[0][kt], false, b, (short)0, acc0, false, false);
            acc1 = __builtin_amdgcn_wmma_f32_16x16x32_bf16(
                false, afrag[1][kt], false, b, (short)0, acc1, false, false);
        }
#pragma unroll
        for (int r = 0; r < 8; ++r) {
            atomicAdd(&agg[doff0[r] + nt * 16], acc0[r]);
            atomicAdd(&agg[doff1[r] + nt * 16], acc1[r]);
        }
    }
}

// ---------------- node GEMM + epilogue ----------------
// hout = relu(h @ Ws + agg + b) stored bf16. 4 waves/block, 32 nodes/wave.
template<int DIN, int DOUT>
__global__ __launch_bounds__(128)
void k_node(const bf16_t* __restrict__ h,    // [N, DIN]
            const bf16_t* __restrict__ wsT,  // [DOUT, DIN] (transposed)
            const float*  __restrict__ agg,  // [N, DOUT]
            const float*  __restrict__ bias, // [DOUT]
            bf16_t*       __restrict__ hout, // [N, DOUT]
            int N)
{
    const int tid  = threadIdx.x;
    const int wave = tid >> 5;
    const int lane = tid & 31;
    const int n0   = blockIdx.x * 128 + wave * 32;
    if (n0 >= N) return;                 // N % 16 == 0 -> 16-row granularity
    const bool has2 = (n0 + 16) < N;     // second subtile validity (N % 32 == 16 tail)

    constexpr int KT = DIN / 32;
    const int mrow = lane & 15;
    const int hi   = lane >> 4;

    // A fragments straight from global (L2-resident), 16B runs.
    // Clamp second-subtile rows so tail reads stay in-bounds (results discarded).
    const int rowA0 = n0 + mrow;
    const int rowA1 = has2 ? (n0 + 16 + mrow) : rowA0;
    v16bf afrag[2][KT];
    const bf16_t* pa0 = h + rowA0 * DIN + hi * 8;
    const bf16_t* pa1 = h + rowA1 * DIN + hi * 8;
#pragma unroll
    for (int kt = 0; kt < KT; ++kt) {
        afrag[0][kt] = make_frag(pa0 + kt * 32, pa0 + kt * 32 + 16);
        afrag[1][kt] = make_frag(pa1 + kt * 32, pa1 + kt * 32 + 16);
    }

    const int obase0 = (n0 + hi * 8) * DOUT + mrow;       // element offset, 32-bit
    const int obase1 = obase0 + 16 * DOUT;

#pragma unroll
    for (int nt = 0; nt < DOUT / 16; ++nt) {
        const int ncol = nt * 16 + mrow;
        const bf16_t* pb = wsT + ncol * DIN + hi * 8;
        const float bv = bias[ncol];
        v8f acc0 = {}, acc1 = {};
#pragma unroll
        for (int kt = 0; kt < KT; ++kt) {
            v16bf b = make_frag(pb + kt * 32, pb + kt * 32 + 16);
            acc0 = __builtin_amdgcn_wmma_f32_16x16x32_bf16(
                false, afrag[0][kt], false, b, (short)0, acc0, false, false);
            acc1 = __builtin_amdgcn_wmma_f32_16x16x32_bf16(
                false, afrag[1][kt], false, b, (short)0, acc1, false, false);
        }
#pragma unroll
        for (int r = 0; r < 8; ++r) {
            const int o0 = obase0 + r * DOUT + nt * 16;
            float v0 = acc0[r] + agg[o0] + bv;
            v0 = v0 > 0.0f ? v0 : 0.0f;  // relu every layer: head consumes relu(h)
            hout[o0] = (bf16_t)v0;
            if (has2) {
                const int o1 = obase1 + r * DOUT + nt * 16;
                float v1 = acc1[r] + agg[o1] + bv;
                v1 = v1 > 0.0f ? v1 : 0.0f;
                hout[o1] = (bf16_t)v1;
            }
        }
    }
}

// ---------------- final head: pred = relu-feat @ Wc + bc ----------------
__global__ void k_final(const bf16_t* __restrict__ feat, const float* __restrict__ wc,
                        const float* __restrict__ bc, float* __restrict__ out, int N)
{
    int n = blockIdx.x * blockDim.x + threadIdx.x;
    if (n >= N) return;
    float s = bc[0];
#pragma unroll 8
    for (int k = 0; k < 128; ++k) s += (float)feat[n * 128 + k] * wc[k];
    out[n] = s;
}

// ---------------- launch ----------------

extern "C" void kernel_launch(void* const* d_in, const int* in_sizes, int n_in,
                              void* d_out, int out_size, void* d_ws, size_t ws_size,
                              hipStream_t stream) {
    (void)in_sizes; (void)n_in; (void)out_size; (void)ws_size;

    const float* x   = (const float*)d_in[0];
    const float* ea  = (const float*)d_in[1];
    const int*   src = (const int*)d_in[2];
    const int*   dst = (const int*)d_in[3];
    const float *Ws[5], *Wm[5], *Bi[5];
    for (int i = 0; i < 5; ++i) {
        Ws[i] = (const float*)d_in[4 + 3 * i];
        Wm[i] = (const float*)d_in[5 + 3 * i];
        Bi[i] = (const float*)d_in[6 + 3 * i];
    }
    const float* Wc = (const float*)d_in[19];
    const float* bc = (const float*)d_in[20];
    float* out = (float*)d_out;

    const int N = 50000, E = 800000;
    const int dins[5]  = {32, 64, 128, 128, 128};
    const int douts[5] = {64, 128, 128, 128, 128};
    const int kpads[5] = {64, 96, 160, 160, 160};

    // workspace carve-up
    char* wsp = (char*)d_ws;
    size_t off = 0;
    auto carve = [&](size_t bytes) -> void* {
        void* p = wsp + off;
        off = (off + bytes + 255) & ~(size_t)255;
        return p;
    };
    bf16_t* hA  = (bf16_t*)carve((size_t)N * 128 * 2);
    bf16_t* hB  = (bf16_t*)carve((size_t)N * 128 * 2);
    float*  agg = (float*) carve((size_t)N * 128 * 4);
    bf16_t* eab = (bf16_t*)carve((size_t)E * 8 * 2);
    bf16_t *wsT[5], *wmT[5];
    for (int i = 0; i < 5; ++i) {
        wsT[i] = (bf16_t*)carve((size_t)douts[i] * dins[i] * 2);   // [DOUT, DIN]
        wmT[i] = (bf16_t*)carve((size_t)douts[i] * kpads[i] * 2);  // [DOUT, KPAD]
    }

    // conversions (re-run every call; deterministic)
    {
        int n = N * 32;
        k_cvt_bf16<<<(n + 255) / 256, 256, 0, stream>>>(x, hA, n);
        n = E * 8;
        k_cvt_bf16<<<(n + 255) / 256, 256, 0, stream>>>(ea, eab, n);
        for (int i = 0; i < 5; ++i) {
            int ns = dins[i] * douts[i];
            k_cvt_transpose<<<(ns + 255) / 256, 256, 0, stream>>>(Ws[i], wsT[i],
                                                                  dins[i], douts[i]);
            int nm = douts[i] * kpads[i];
            k_cvt_transpose_pad<<<(nm + 255) / 256, 256, 0, stream>>>(
                Wm[i], wmT[i], dins[i] + 8, kpads[i], douts[i]);
        }
    }

    const int egrid = E / 64;            // 64 edges per block (2 waves x 32)
    const int ngrid = (N + 127) / 128;   // 128 nodes per block (4 waves x 32)
    bf16_t* hin = hA;
    bf16_t* hout = hB;

    for (int i = 0; i < 5; ++i) {
        const int na = N * douts[i];
        k_zero_f32<<<(na + 255) / 256, 256, 0, stream>>>(agg, na);
        switch (i) {
        case 0:
            k_edge<32, 64, 64><<<egrid, 64, 0, stream>>>(hin, eab, src, dst, wmT[i], agg);
            k_node<32, 64><<<ngrid, 128, 0, stream>>>(hin, wsT[i], agg, Bi[i], hout, N);
            break;
        case 1:
            k_edge<64, 128, 96><<<egrid, 64, 0, stream>>>(hin, eab, src, dst, wmT[i], agg);
            k_node<64, 128><<<ngrid, 128, 0, stream>>>(hin, wsT[i], agg, Bi[i], hout, N);
            break;
        default:
            k_edge<128, 128, 160><<<egrid, 64, 0, stream>>>(hin, eab, src, dst, wmT[i], agg);
            k_node<128, 128><<<ngrid, 128, 0, stream>>>(hin, wsT[i], agg, Bi[i], hout, N);
            break;
        }
        bf16_t* t = hin; hin = hout; hout = t;
    }
    k_final<<<(N + 255) / 256, 256, 0, stream>>>(hin, Wc, bc, out, N);
}
